// ResidualVectorQuantizer_25288767439192
// MI455X (gfx1250) — matrix operand order repaired and checked by reading, not compiled, not run on hardware
//
#include <hip/hip_runtime.h>
#include <hip/hip_bf16.h>
#include <hip/hip_fp16.h>

// ---------------- problem constants ----------------
#define EMBED_DIM 512
#define CBK       2048
#define NLAYERS   4
#define NTOK      32768
#define COMMIT    0.25f
// softmax temperature scale = sqrt(512); we use 1/scale
#define INV_SCALE 0.04419417382415922f

// ---------------- tiling ----------------
#define MTILE  32                 // tokens per workgroup
#define WAVES  8                  // 256 threads, wave32
#define KSLICE (CBK / WAVES)      // 256 codes per wave
#define KGRP   4                  // k-subtiles (of 16) sharing one A-fragment load

typedef __attribute__((ext_vector_type(16))) __bf16 v16bf;
typedef __attribute__((ext_vector_type(8)))  __bf16 v8bf;
typedef __attribute__((ext_vector_type(8)))  float  v8f;

union FragAB { v16bf v; struct { v8bf lo, hi; } p; };

__device__ __forceinline__ unsigned short f32_to_bf16(float f) {
  unsigned u = __float_as_uint(f);
  unsigned r = u + 0x7FFFu + ((u >> 16) & 1u);   // round-to-nearest-even
  return (unsigned short)(r >> 16);
}

// ---------------- prep: zero loss + usage ----------------
__global__ void rvq_zero(float* __restrict__ loss, float* __restrict__ usage) {
  int i = blockIdx.x * blockDim.x + threadIdx.x;
  if (i == 0) *loss = 0.0f;
  if (i < NLAYERS * CBK) usage[i] = 0.0f;
}

// ---------------- prep: codebook fp32 -> bf16 + row norms ----------------
// one wave per codebook row (L*K rows)
__global__ void rvq_cb_prep(const float* __restrict__ cb,
                            unsigned short* __restrict__ cbbf,
                            float* __restrict__ cbnorm) {
  int row  = (blockIdx.x * blockDim.x + threadIdx.x) >> 5;
  int lane = threadIdx.x & 31;
  if (row >= NLAYERS * CBK) return;
  const float* src = cb + (size_t)row * EMBED_DIM;
  unsigned short* dst = cbbf + (size_t)row * EMBED_DIM;
  float s = 0.0f;
  for (int d = lane; d < EMBED_DIM; d += 32) {
    float v = src[d];
    s += v * v;
    dst[d] = f32_to_bf16(v);
  }
  for (int sh = 16; sh; sh >>= 1) s += __shfl_xor(s, sh, 32);
  if (lane == 0) cbnorm[row] = s;
}

// ---------------- main per-layer kernel ----------------
__global__ __launch_bounds__(256)
void rvq_layer(const float* __restrict__ cb32,          // [L][K][D] fp32
               const unsigned short* __restrict__ cbbf, // [L][K][D] bf16
               const float* __restrict__ cbnorm,        // [L][K]
               float* __restrict__ residual,            // [N][D] fp32 (ws, updated)
               float* __restrict__ qsum,                // d_out [N][D]
               float* __restrict__ lossAcc,             // d_out scalar
               float* __restrict__ codesOut,            // d_out [N][L] (stored as float)
               float* __restrict__ usageOut,            // d_out [L][K]
               int layer)
{
  constexpr int D = EMBED_DIM;
  constexpr int K = CBK;
  const int tid  = threadIdx.x;
  const int lane = tid & 31;
  const int wave = tid >> 5;
  const int tok0 = blockIdx.x * MTILE;

  extern __shared__ char smem_raw[];
  unsigned short* Abuf = (unsigned short*)smem_raw;              // [MTILE][D] bf16 (32 KB)
  _Float16*       Dbuf = (_Float16*)(Abuf + MTILE * D);          // [MTILE][K] fp16 (128 KB)
  float* RNORM = (float*)(Dbuf + (size_t)MTILE * K);             // [MTILE]
  float* MNP   = RNORM + MTILE;                                  // [WAVES][MTILE]
  int*   AGP   = (int*)(MNP + WAVES * MTILE);                    // [WAVES][MTILE]
  float* DMIN  = (float*)(AGP + WAVES * MTILE);                  // [MTILE]
  int*   ARG   = (int*)(DMIN + MTILE);                           // [MTILE]
  float* INVZ  = (float*)(ARG + MTILE);                          // [MTILE]

  const unsigned short* cbbfL = cbbf + (size_t)layer * K * D;
  const float* cbnormL = cbnorm + (size_t)layer * K;
  const float* cb32L   = cb32 + (size_t)layer * K * D;

  // ---- Phase 0: residual tile -> bf16 LDS, per-token norms ----
  for (int s = 0; s < MTILE / WAVES; ++s) {
    int m = wave * (MTILE / WAVES) + s;
    const float* rrow = residual + (size_t)(tok0 + m) * D;
    float acc = 0.0f;
    for (int d = lane; d < D; d += 32) {
      float v = rrow[d];
      acc += v * v;
      Abuf[m * D + d] = f32_to_bf16(v);
    }
    for (int sh = 16; sh; sh >>= 1) acc += __shfl_xor(acc, sh, 32);
    if (lane == 0) RNORM[m] = acc;
  }
  __syncthreads();

  // ---- Phase A: WMMA distance GEMM over this wave's 256-code slice ----
  const int nlane = lane & 15;
  const int hi    = lane >> 4;            // half-wave select
  const int aoff  = hi ? 8 : 0;           // A-frag K-run offset (ISA 16-bit A layout)
  const int boff  = hi * 8;               // B-frag K(d)-run offset (matches A)

  float dmin[16];
  int   amin[16];
#pragma unroll
  for (int i = 0; i < 16; ++i) { dmin[i] = 3.4e38f; amin[i] = 0; }

  const int kbase = wave * KSLICE;
  for (int kt = 0; kt < KSLICE; kt += 16 * KGRP) {
    int kcol[KGRP];
    float cn[KGRP];
    const unsigned short* brow[KGRP];
    v8f acc0[KGRP] = {};   // token rows 0..15
    v8f acc1[KGRP] = {};   // token rows 16..31
#pragma unroll
    for (int j = 0; j < KGRP; ++j) {
      kcol[j] = kbase + kt + j * 16 + nlane;     // this lane's code column
      cn[j]   = cbnormL[kcol[j]];
      brow[j] = cbbfL + (size_t)kcol[j] * D;
    }

#pragma unroll 2
    for (int d0 = 0; d0 < D; d0 += 32) {
      // A frags (one LDS fetch reused across KGRP wmma columns):
      // row = nlane, K-runs {aoff..aoff+7, aoff+16..aoff+23} within the 32-chunk
      const unsigned short* ar0 = Abuf + (size_t)nlane * D + d0 + aoff;
      const unsigned short* ar1 = Abuf + (size_t)(16 + nlane) * D + d0 + aoff;
      FragAB a0, a1;
      a0.p.lo = *(const v8bf*)(ar0);
      a0.p.hi = *(const v8bf*)(ar0 + 16);
      a1.p.lo = *(const v8bf*)(ar1);
      a1.p.hi = *(const v8bf*)(ar1 + 16);

#pragma unroll
      for (int j = 0; j < KGRP; ++j) {
        // B frag: 16-bit B 32x16 layout — lane column = nlane,
        // d-runs {boff..boff+7, boff+16..boff+23}, same as A
        FragAB b;
        b.p.lo = *(const v8bf*)(brow[j] + d0 + boff);
        b.p.hi = *(const v8bf*)(brow[j] + d0 + boff + 16);
        acc0[j] = __builtin_amdgcn_wmma_f32_16x16x32_bf16(false, a0.v, false, b.v,
                                                          (short)0, acc0[j], false, false);
        acc1[j] = __builtin_amdgcn_wmma_f32_16x16x32_bf16(false, a1.v, false, b.v,
                                                          (short)0, acc1[j], false, false);
      }
    }

    // dist = |r|^2 + |c|^2 - 2 r.c ; spill as fp16, track fp32 argmin
#pragma unroll
    for (int j = 0; j < KGRP; ++j) {
#pragma unroll
      for (int i = 0; i < 8; ++i) {
        int m0 = hi * 8 + i;                     // C/D layout: VGPR i, halves hold M=i / M=8+i
        float d0v = RNORM[m0] + cn[j] - 2.0f * acc0[j][i];
        Dbuf[(size_t)m0 * K + kcol[j]] = (_Float16)d0v;
        if (d0v < dmin[i]) { dmin[i] = d0v; amin[i] = kcol[j]; }
        int m1 = 16 + hi * 8 + i;
        float d1v = RNORM[m1] + cn[j] - 2.0f * acc1[j][i];
        Dbuf[(size_t)m1 * K + kcol[j]] = (_Float16)d1v;
        if (d1v < dmin[8 + i]) { dmin[8 + i] = d1v; amin[8 + i] = kcol[j]; }
      }
    }
  }

  // reduce argmin across the 16 lanes of each half-wave (lowest-index tie-break)
#pragma unroll
  for (int i = 0; i < 16; ++i) {
    float dv = dmin[i]; int av = amin[i];
    for (int sh = 1; sh < 16; sh <<= 1) {
      float od = __shfl_xor(dv, sh, 32);
      int   oa = __shfl_xor(av, sh, 32);
      if (od < dv || (od == dv && oa < av)) { dv = od; av = oa; }
    }
    if (nlane == 0) {
      int m = (i < 8) ? (hi * 8 + i) : (16 + hi * 8 + (i - 8));
      MNP[wave * MTILE + m] = dv;
      AGP[wave * MTILE + m] = av;
    }
  }
  __syncthreads();

  // ---- Phase B: combine across waves, emit codes ----
  if (tid < MTILE) {
    float best = 3.4e38f; int barg = 0;
    for (int w = 0; w < WAVES; ++w) {           // ascending k-slices -> correct tie-break
      float dv = MNP[w * MTILE + tid];
      int   av = AGP[w * MTILE + tid];
      if (dv < best || (dv == best && av < barg)) { best = dv; barg = av; }
    }
    DMIN[tid] = best;
    ARG[tid]  = barg;
    codesOut[(size_t)(tok0 + tid) * NLAYERS + layer] = (float)barg;
  }
  __syncthreads();

  // ---- Phase C: fp32 gather, residual update, loss; softmax normalizer Z ----
  float lsum = 0.0f;
  for (int s = 0; s < MTILE / WAVES; ++s) {
    int m = wave * (MTILE / WAVES) + s;
    int code = ARG[m];
    const float* qrow = cb32L + (size_t)code * D;
    float* rrow = residual + (size_t)(tok0 + m) * D;
    float* qs   = qsum + (size_t)(tok0 + m) * D;
    for (int d = lane; d < D; d += 32) {
      float q  = qrow[d];
      float nr = rrow[d] - q;
      rrow[d] = nr;                 // residual for next layer
      lsum += nr * nr;              // commitment loss (diff == new residual)
      if (layer == 0) qs[d] = q; else qs[d] += q;
    }
    float z = 0.0f;
    float dm = DMIN[m];
    for (int k = lane; k < K; k += 32) {
      float dist = (float)Dbuf[(size_t)m * K + k];
      z += __expf((dm - dist) * INV_SCALE);     // max-shifted, each term <= ~1
    }
    for (int sh = 16; sh; sh >>= 1) z += __shfl_xor(z, sh, 32);
    if (lane == 0) INVZ[m] = 1.0f / z;
  }
  for (int sh = 16; sh; sh >>= 1) lsum += __shfl_xor(lsum, sh, 32);
  if (lane == 0) atomicAdd(lossAcc, lsum * (COMMIT / ((float)NTOK * (float)D)));
  __syncthreads();

  // ---- Phase D: soft usage accumulation ----
  for (int k = tid; k < K; k += 256) {
    float s = 0.0f;
#pragma unroll 8
    for (int m = 0; m < MTILE; ++m) {
      float dist = (float)Dbuf[(size_t)m * K + k];
      s += __expf((DMIN[m] - dist) * INV_SCALE) * INVZ[m];
    }
    atomicAdd(usageOut + (size_t)layer * K + k, s * (1.0f / (float)NTOK));
  }
}

// ---------------- host-side launch ----------------
extern "C" void kernel_launch(void* const* d_in, const int* in_sizes, int n_in,
                              void* d_out, int out_size, void* d_ws, size_t ws_size,
                              hipStream_t stream) {
  (void)in_sizes; (void)n_in; (void)out_size; (void)ws_size;
  const float* inputs    = (const float*)d_in[0];   // [N, D]
  const float* codebooks = (const float*)d_in[1];   // [L, K, D]

  // workspace carve: residual (64 MB) | bf16 codebooks (8 MB) | code norms (32 KB)
  char* ws = (char*)d_ws;
  float*          residual = (float*)ws;
  size_t off = (size_t)NTOK * EMBED_DIM * sizeof(float);
  unsigned short* cbbf     = (unsigned short*)(ws + off);
  off += (size_t)NLAYERS * CBK * EMBED_DIM * sizeof(unsigned short);
  float*          cbnorm   = (float*)(ws + off);

  // output carve: quantized_st [N*D] | loss [1] | codes [N*L] | soft_usage [L*K]
  float* qsum  = (float*)d_out;
  float* loss  = qsum + (size_t)NTOK * EMBED_DIM;
  float* codes = loss + 1;
  float* usage = codes + (size_t)NTOK * NLAYERS;

  // residual <- inputs
  hipMemcpyAsync(residual, inputs, (size_t)NTOK * EMBED_DIM * sizeof(float),
                 hipMemcpyDeviceToDevice, stream);
  rvq_zero<<<(NLAYERS * CBK + 255) / 256, 256, 0, stream>>>(loss, usage);
  rvq_cb_prep<<<(NLAYERS * CBK) / 8, 256, 0, stream>>>(codebooks, cbbf, cbnorm);

  // dynamic LDS: A tile (32KB) + dist buffer (128KB) + reductions
  size_t smem = (size_t)MTILE * EMBED_DIM * 2          // Abuf
              + (size_t)MTILE * CBK * 2                // Dbuf
              + (size_t)MTILE * 4                      // RNORM
              + (size_t)WAVES * MTILE * 4 * 2          // MNP + AGP
              + (size_t)MTILE * 4 * 3;                 // DMIN + ARG + INVZ

  for (int l = 0; l < NLAYERS; ++l) {
    rvq_layer<<<NTOK / MTILE, 256, smem, stream>>>(
        codebooks, cbbf, cbnorm, residual, qsum, loss, codes, usage, l);
  }
}
